// FFB6D_39951785787552
// MI455X (gfx1250) — matrix operand order repaired and checked by reading, not compiled, not run on hardware
//
#include <hip/hip_runtime.h>

typedef __bf16 bf16;
typedef __attribute__((ext_vector_type(16))) __bf16 v16bf;
typedef __attribute__((ext_vector_type(8)))  __bf16 v8bf;
typedef __attribute__((ext_vector_type(8)))  float  v8f;
typedef int i32x4 __attribute__((vector_size(16)));

#if defined(__has_builtin)
#  if __has_builtin(__builtin_amdgcn_global_load_async_to_lds_b128) && \
      __has_builtin(__builtin_amdgcn_s_wait_asynccnt)
#    define FFB_ASYNC 1
#  endif
#endif

#define GSTR 40  // padded LDS K-stride in bf16 elems: 80B rows, 16B-aligned chunks

// ---------------------------------------------------------------------------
// f32 -> bf16 conversion
// ---------------------------------------------------------------------------
__global__ __launch_bounds__(256)
void k_cvt(const float* __restrict__ s, bf16* __restrict__ d, long long n) {
  long long t = (long long)blockIdx.x * 256 + threadIdx.x;
  if (t < n) d[t] = (bf16)s[t];
}

// ---------------------------------------------------------------------------
// im2col for the 7x7/2 pad3 stem: col[k][n], k = ic*49+kh*7+kw (147), n = oh*320+ow
// ---------------------------------------------------------------------------
__global__ __launch_bounds__(256)
void k_im2col_stem(const float* __restrict__ rgb, bf16* __restrict__ col) {
  long long t = (long long)blockIdx.x * 256 + threadIdx.x;
  const long long total = (long long)147 * 76800;
  if (t >= total) return;
  int n = (int)(t % 76800);
  int k = (int)(t / 76800);
  int ow = n % 320, oh = n / 320;
  int kw = k % 7, kh = (k / 7) % 7, ic = k / 49;
  int iw = ow * 2 - 3 + kw, ih = oh * 2 - 3 + kh;
  float v = 0.f;
  if (iw >= 0 && iw < 640 && ih >= 0 && ih < 480)
    v = rgb[((size_t)ic * 480 + ih) * 640 + iw];
  col[t] = (bf16)v;
}

// ---------------------------------------------------------------------------
// 3x3 stride-2 pad-1 max pool, NCHW bf16
// ---------------------------------------------------------------------------
__global__ __launch_bounds__(256)
void k_maxpool3(const bf16* __restrict__ in, bf16* __restrict__ out,
                int C, int Hi, int Wi) {
  int Ho = Hi / 2, Wo = Wi / 2;
  long long t = (long long)blockIdx.x * 256 + threadIdx.x;
  if (t >= (long long)C * Ho * Wo) return;
  int wo = (int)(t % Wo);
  int ho = (int)((t / Wo) % Ho);
  int c  = (int)(t / ((long long)Wo * Ho));
  float best = -3.4e38f;
  for (int kh = 0; kh < 3; ++kh)
    for (int kw = 0; kw < 3; ++kw) {
      int h = ho * 2 - 1 + kh, w = wo * 2 - 1 + kw;
      if (h >= 0 && h < Hi && w >= 0 && w < Wi) {
        float v = (float)in[((size_t)c * Hi + h) * Wi + w];
        best = v > best ? v : best;
      }
    }
  out[t] = (bf16)best;
}

// ---------------------------------------------------------------------------
// nearest-neighbor x2 upsample, NCHW bf16
// ---------------------------------------------------------------------------
__global__ __launch_bounds__(256)
void k_up2(const bf16* __restrict__ in, bf16* __restrict__ out,
           int C, int H, int W) {
  int H2 = H * 2, W2 = W * 2;
  long long t = (long long)blockIdx.x * 256 + threadIdx.x;
  if (t >= (long long)C * H2 * W2) return;
  int wo = (int)(t % W2);
  int ho = (int)((t / W2) % H2);
  int c  = (int)(t / ((long long)W2 * H2));
  out[t] = in[((size_t)c * H + (ho >> 1)) * W + (wo >> 1)];
}

// ---------------------------------------------------------------------------
// stride-2 spatial slice x[:, ::2, ::2]
// ---------------------------------------------------------------------------
__global__ __launch_bounds__(256)
void k_slice2(const bf16* __restrict__ in, bf16* __restrict__ out,
              int C, int H, int W) {
  int Ho = H / 2, Wo = W / 2;
  long long t = (long long)blockIdx.x * 256 + threadIdx.x;
  if (t >= (long long)C * Ho * Wo) return;
  int wo = (int)(t % Wo);
  int ho = (int)((t / Wo) % Ho);
  int c  = (int)(t / ((long long)Wo * Ho));
  out[t] = in[((size_t)c * H + ho * 2) * W + wo * 2];
}

// ---------------------------------------------------------------------------
// gather + max over Kn neighbors: out[c,m] = max_k feat[c, idx[m*Kn+k]]
// (Kn==1 == nearest_interp / choose-gather). Pure random-access bandwidth.
// ---------------------------------------------------------------------------
__global__ __launch_bounds__(256)
void k_gather_max(const bf16* __restrict__ feat, int C, int Ns,
                  const int* __restrict__ idx, int M, int Kn,
                  bf16* __restrict__ out) {
  long long t = (long long)blockIdx.x * 256 + threadIdx.x;
  if (t >= (long long)C * M) return;
  int m = (int)(t % M);
  int c = (int)(t / M);
  const bf16* f = feat + (size_t)c * Ns;
  const int* id = idx + (size_t)m * Kn;
  float best = -3.4e38f;
  for (int k = 0; k < Kn; ++k) {
    float v = (float)f[id[k]];
    best = v > best ? v : best;
  }
  out[(size_t)c * M + m] = (bf16)best;
}

// ---------------------------------------------------------------------------
// Fused GEMM: Y = act( gamma ∘ (A[MxK] * B[KxN]) + beta ), bf16 in, WMMA bf16.
// 64x64 block tile, 256 threads = 8 wave32 waves, wave tile 16x32 (2 WMMAs),
// K-step 32, double-buffered LDS. Tile-uniform fast/slow staging paths:
// interior tiles use one async b128 global->LDS copy per thread for A and one
// coalesced 16B vector load (+ transpose scatter) per thread for B; only true
// edge tiles (ragged M/N/K) run the guarded scalar zero-padding path.
// layout: 0=bf16 [M,N], 1=f32 [M,N], 2=f32 grouped-3 transpose (kp/ctr heads).
// ---------------------------------------------------------------------------
__global__ __launch_bounds__(256)
void k_gemm(const bf16* __restrict__ A, const bf16* __restrict__ B,
            const float* __restrict__ gamma, const float* __restrict__ beta,
            bf16* __restrict__ Obf, float* __restrict__ Of32,
            int M, int N, int K, int relu, int layout) {
  __shared__ bf16 As[2][64 * GSTR];
  __shared__ bf16 Bs[2][64 * GSTR];

  const int tid  = threadIdx.x;
  const int m0   = blockIdx.y * 64;
  const int n0   = blockIdx.x * 64;
  const int wave = tid >> 5;
  const int lane = tid & 31;
  const int wm   = wave >> 1;  // 0..3: 16-row strip
  const int wn   = wave & 1;   // 0..1: 32-col strip
  const int half = lane >> 4;
  const int l16  = lane & 15;

  v8f acc0, acc1;
#pragma unroll
  for (int j = 0; j < 8; ++j) { acc0[j] = 0.f; acc1[j] = 0.f; }

  const int ktiles = (K + 31) >> 5;
  // block-uniform fast-path eligibility
  const bool fastA = (m0 + 64 <= M) && ((K & 7) == 0);  // rows in range, rows 16B-aligned
  const bool fastB = (n0 + 64 <= N) && ((N & 7) == 0);  // cols in range, rows 16B-aligned

  auto stage = [&](int kt, int buf) {
    const int kbase = kt << 5;
    const bool fullK = (kbase + 32 <= K);
    // ---- A tile 64x32: thread -> 16B of one weight row ----
    {
      const int row = tid >> 2, cb = (tid & 3) * 8;
      const int m = m0 + row;
      const int kb = kbase + cb;
      if (fastA && fullK) {
#ifdef FFB_ASYNC
        __builtin_amdgcn_global_load_async_to_lds_b128(
            (__attribute__((address_space(1))) i32x4*)(A + (size_t)m * K + kb),
            (__attribute__((address_space(3))) i32x4*)&As[buf][row * GSTR + cb],
            0, 0);
#else
        *(v8bf*)&As[buf][row * GSTR + cb] = *(const v8bf*)(A + (size_t)m * K + kb);
#endif
      } else {
#pragma unroll
        for (int j = 0; j < 8; ++j) {
          const int k = kb + j;
          As[buf][row * GSTR + cb + j] =
              (m < M && k < K) ? A[(size_t)m * K + k] : (bf16)0.f;
        }
      }
    }
    // ---- B tile 32x64 -> LDS transposed [n][k] ----
    if (fastB && fullK) {
      // thread: row k = kbase + (tid>>3), 16B chunk of 8 cols at nc
      const int kl = tid >> 3;
      const int nc = (tid & 7) * 8;
      v8bf v = *(const v8bf*)(B + (size_t)(kbase + kl) * N + n0 + nc);
#pragma unroll
      for (int j = 0; j < 8; ++j)
        Bs[buf][(nc + j) * GSTR + kl] = v[j];
    } else {
      const int n = tid & 63, kb = (tid >> 6) * 8;
      const int nn = n0 + n;
#pragma unroll
      for (int j = 0; j < 8; ++j) {
        const int k = kbase + kb + j;
        Bs[buf][n * GSTR + kb + j] =
            (nn < N && k < K) ? B[(size_t)k * N + nn] : (bf16)0.f;
      }
    }
  };

  auto compute = [&](int buf) {
    // A frag (ISA 16-bit A 16x32): lane row = l16; K chunks {b..b+7, b+16..b+23}, b=8*half
    const bf16* ap = &As[buf][(wm * 16 + l16) * GSTR + half * 8];
    v8bf alo = *(const v8bf*)(ap);
    v8bf ahi = *(const v8bf*)(ap + 16);
    v16bf a;
#pragma unroll
    for (int j = 0; j < 8; ++j) { a[j] = alo[j]; a[8 + j] = ahi[j]; }
    // B frags (ISA 16-bit B): lane col = l16; K chunks {b..b+7, b+16..b+23}, b=8*half
    const bf16* bp0 = &Bs[buf][(wn * 32 + l16) * GSTR + half * 8];
    const bf16* bp1 = &Bs[buf][(wn * 32 + 16 + l16) * GSTR + half * 8];
    v8bf x0 = *(const v8bf*)(bp0), x1 = *(const v8bf*)(bp0 + 16);
    v8bf y0 = *(const v8bf*)(bp1), y1 = *(const v8bf*)(bp1 + 16);
    v16bf b0, b1;
#pragma unroll
    for (int j = 0; j < 8; ++j) {
      b0[j] = x0[j]; b0[8 + j] = x1[j];
      b1[j] = y0[j]; b1[8 + j] = y1[j];
    }
    acc0 = __builtin_amdgcn_wmma_f32_16x16x32_bf16(false, a, false, b0,
                                                   (short)0, acc0, false, false);
    acc1 = __builtin_amdgcn_wmma_f32_16x16x32_bf16(false, a, false, b1,
                                                   (short)0, acc1, false, false);
  };

  stage(0, 0);
#ifdef FFB_ASYNC
  if (fastA) __builtin_amdgcn_s_wait_asynccnt(0);
#endif
  __syncthreads();
  for (int kt = 0; kt < ktiles; ++kt) {
    const int buf = kt & 1;
    if (kt + 1 < ktiles) stage(kt + 1, buf ^ 1);
    compute(buf);
#ifdef FFB_ASYNC
    if (fastA) __builtin_amdgcn_s_wait_asynccnt(0);
#endif
    __syncthreads();
  }

  // Epilogue (ISA C/D layout: VGPR r -> M = r + 8*half, N = l16)
  const int mb = m0 + wm * 16 + half * 8;
  const int nb = n0 + wn * 32 + l16;
#pragma unroll
  for (int r = 0; r < 8; ++r) {
    const int m = mb + r;
    if (m >= M) continue;
    const float gsc = gamma ? gamma[m] : 1.f;
    const float bsh = beta ? beta[m] : 0.f;
#pragma unroll
    for (int p = 0; p < 2; ++p) {
      const int n = nb + p * 16;
      if (n >= N) continue;
      float v = (p ? acc1[r] : acc0[r]) * gsc + bsh;
      if (relu && v < 0.f) v = 0.f;
      if (layout == 0)
        Obf[(size_t)m * N + n] = (bf16)v;
      else if (layout == 1)
        Of32[(size_t)m * N + n] = v;
      else
        Of32[(size_t)(m / 3) * ((size_t)N * 3) + (size_t)n * 3 + (m % 3)] = v;
    }
  }
}

// ---------------------------------------------------------------------------
// Host orchestration
// ---------------------------------------------------------------------------
extern "C" void kernel_launch(void* const* d_in, const int* in_sizes, int n_in,
                              void* d_out, int out_size, void* d_ws, size_t ws_size,
                              hipStream_t stream) {
  (void)in_sizes; (void)n_in; (void)out_size; (void)ws_size;

  // ---- fixed tensor inputs (setup_inputs insertion order) ----
  const float* rgb    = (const float*)d_in[0];
  const float* cld    = (const float*)d_in[1];
  const int*   choose = (const int*)d_in[2];
  const int *cld_nei[4], *cld_sub[4], *r2p_ds[4], *p2r_ds[4], *cld_interp[4];
  for (int i = 0; i < 4; ++i) {
    cld_nei[i]    = (const int*)d_in[3 + i * 5 + 0];
    cld_sub[i]    = (const int*)d_in[3 + i * 5 + 1];
    r2p_ds[i]     = (const int*)d_in[3 + i * 5 + 2];
    p2r_ds[i]     = (const int*)d_in[3 + i * 5 + 3];
    cld_interp[i] = (const int*)d_in[3 + i * 5 + 4];
  }
  const int *p2r_up[3], *r2p_up[3];
  for (int i = 0; i < 3; ++i) {
    p2r_up[i] = (const int*)d_in[23 + 2 * i];
    r2p_up[i] = (const int*)d_in[24 + 2 * i];
  }

  // ---- params: pytree leaves from d_in[29], key-sorted; CBR leaves b,g,w ----
  int pi = 29;
  auto nf = [&]() -> const float* { return (const float*)d_in[pi++]; };
  struct CBRp { const float* b; const float* g; const float* w; int o; int i; bf16* wb; };
  auto getc = [&](int o, int i) -> CBRp {
    CBRp c; c.b = nf(); c.g = nf(); c.w = nf(); c.o = o; c.i = i; c.wb = nullptr; return c;
  };
  CBRp cnn_ds[4]      = { getc(64,64),  getc(128,64),  getc(512,128),  getc(1024,512) };
  CBRp cnn_final      =   getc(64,64);
  CBRp cnn_up1        =   getc(256,1024);
  CBRp cnn_up2        =   getc(64,256);
  CBRp cnn_up3        =   getc(64,64);
  CBRp ctr_h[3]       = { getc(128,128), getc(128,128), getc(128,128) };
  const float* ctr_bias = nf(); const float* ctr_w = nf();
  CBRp ds_p2r_fuse[4] = { getc(64,128),  getc(128,256), getc(512,1024), getc(1024,2048) };
  CBRp ds_p2r_pre[4]  = { getc(64,64),   getc(128,128), getc(512,256),  getc(1024,512) };
  CBRp ds_r2p_fuse[4] = { getc(64,128),  getc(128,256), getc(256,512),  getc(512,1024) };
  CBRp ds_r2p_pre[4]  = { getc(64,64),   getc(128,128), getc(256,512),  getc(512,1024) };
  CBRp fc0            =   getc(8,9);
  CBRp kp_h[3]        = { getc(128,128), getc(128,128), getc(128,128) };
  const float* kp_bias = nf(); const float* kp_w = nf();
  CBRp rndla_ds[4]    = { getc(64,16),   getc(128,128), getc(256,256),  getc(512,512) };
  CBRp rndla_up[4]    = { getc(256,768), getc(128,384), getc(64,192),   getc(64,128) };
  CBRp seg_h[3]       = { getc(128,128), getc(128,128), getc(128,128) };
  const float* seg_bias = nf(); const float* seg_w = nf();
  CBRp stem           =   getc(64,147);
  CBRp up_p2r_fuse[3] = { getc(256,512), getc(64,128),  getc(64,128) };
  CBRp up_p2r_pre[3]  = { getc(256,256), getc(64,128),  getc(64,64) };
  CBRp up_r2p_fuse[3] = { getc(256,512), getc(128,256), getc(64,128) };
  CBRp up_r2p_pre[3]  = { getc(256,256), getc(128,64),  getc(64,64) };

  // ---- workspace bump allocator ----
  char* base = (char*)d_ws;
  size_t cur = 0;
  auto alloc = [&](size_t bytes) -> void* {
    size_t a = (cur + 255) & ~(size_t)255;
    cur = a + bytes;
    return (void*)(base + a);
  };
  auto nblk = [](long long n) { return (unsigned)((n + 255) / 256); };
  auto cvt = [&](const float* s, bf16* d, long long n) {
    k_cvt<<<nblk(n), 256, 0, stream>>>(s, d, n);
  };
  auto gemm = [&](const bf16* W, const bf16* X, const float* g, const float* b,
                  bf16* obf, float* of32, int M, int N, int K, int relu, int layout) {
    dim3 grid((unsigned)((N + 63) / 64), (unsigned)((M + 63) / 64));
    k_gemm<<<grid, 256, 0, stream>>>(W, X, g, b, obf, of32, M, N, K, relu, layout);
  };
  auto cbr = [&](CBRp& c, const bf16* X, int N, bf16* out) {
    gemm(c.wb, X, c.g, c.b, out, nullptr, c.o, N, c.i, 1, 0);
  };
  auto gath = [&](const bf16* feat, int C, int Ns, const int* idx, int M, int Kn, bf16* out) {
    k_gather_max<<<nblk((long long)C * M), 256, 0, stream>>>(feat, C, Ns, idx, M, Kn, out);
  };
  auto d2d = [&](bf16* dst, const bf16* src, long long elems) {
    (void)hipMemcpyAsync(dst, src, (size_t)elems * sizeof(bf16),
                         hipMemcpyDeviceToDevice, stream);
  };

  // ---- convert all weights to bf16 once (persistent) ----
  CBRp* all_cbr[] = {
    &cnn_ds[0], &cnn_ds[1], &cnn_ds[2], &cnn_ds[3], &cnn_final, &cnn_up1, &cnn_up2, &cnn_up3,
    &ctr_h[0], &ctr_h[1], &ctr_h[2],
    &ds_p2r_fuse[0], &ds_p2r_fuse[1], &ds_p2r_fuse[2], &ds_p2r_fuse[3],
    &ds_p2r_pre[0], &ds_p2r_pre[1], &ds_p2r_pre[2], &ds_p2r_pre[3],
    &ds_r2p_fuse[0], &ds_r2p_fuse[1], &ds_r2p_fuse[2], &ds_r2p_fuse[3],
    &ds_r2p_pre[0], &ds_r2p_pre[1], &ds_r2p_pre[2], &ds_r2p_pre[3],
    &fc0, &kp_h[0], &kp_h[1], &kp_h[2],
    &rndla_ds[0], &rndla_ds[1], &rndla_ds[2], &rndla_ds[3],
    &rndla_up[0], &rndla_up[1], &rndla_up[2], &rndla_up[3],
    &seg_h[0], &seg_h[1], &seg_h[2], &stem,
    &up_p2r_fuse[0], &up_p2r_fuse[1], &up_p2r_fuse[2],
    &up_p2r_pre[0], &up_p2r_pre[1], &up_p2r_pre[2],
    &up_r2p_fuse[0], &up_r2p_fuse[1], &up_r2p_fuse[2],
    &up_r2p_pre[0], &up_r2p_pre[1], &up_r2p_pre[2],
  };
  for (CBRp* c : all_cbr) {
    long long n = (long long)c->o * c->i;
    c->wb = (bf16*)alloc((size_t)n * 2);
    cvt(c->w, c->wb, n);
  }
  bf16* seg_wb = (bf16*)alloc(22 * 128 * 2); cvt(seg_w, seg_wb, 22 * 128);
  bf16* kp_wb  = (bf16*)alloc(24 * 128 * 2); cvt(kp_w,  kp_wb,  24 * 128);
  bf16* ctr_wb = (bf16*)alloc(3 * 128 * 2);  cvt(ctr_w, ctr_wb, 3 * 128);

  // ---- persistent activations ----
  const int NP[5]  = {12800, 3200, 800, 200, 50};
  const int DSR[4] = {64, 128, 512, 1024};
  const int DSP[4] = {64, 128, 256, 512};
  bf16* pemb0 = (bf16*)alloc((size_t)8 * 12800 * 2);
  bf16* dse0  = (bf16*)alloc((size_t)64 * 12800 * 2);     // ds_emb[0] (enc)
  bf16* dsp[4];
  for (int i = 0; i < 4; ++i) dsp[i] = (bf16*)alloc((size_t)DSP[i] * NP[i + 1] * 2);
  const int UPP[3] = {256, 128, 64};
  const int UPR[3] = {256, 64, 64};
  bf16* upp[3];
  for (int i = 0; i < 3; ++i) upp[i] = (bf16*)alloc((size_t)UPP[i] * NP[3 - i] * 2);
  bf16* pembF = (bf16*)alloc((size_t)64 * 12800 * 2);
  bf16* rgbA  = (bf16*)alloc((size_t)64 * 307200 * 2);
  bf16* rgbB  = (bf16*)alloc((size_t)64 * 307200 * 2);

  const size_t temp_base = cur;
  auto reset = [&]() { cur = temp_base; };

  // ================= stem + fc0 =================
  reset();
  {
    long long n9 = (long long)9 * 12800;
    bf16* cldb = (bf16*)alloc((size_t)n9 * 2);
    cvt(cld, cldb, n9);
    gemm(fc0.wb, cldb, fc0.g, fc0.b, pemb0, nullptr, 8, 12800, 9, 1, 0);

    bf16* col = (bf16*)alloc((size_t)147 * 76800 * 2);
    k_im2col_stem<<<nblk((long long)147 * 76800), 256, 0, stream>>>(rgb, col);
    bf16* so = (bf16*)alloc((size_t)64 * 76800 * 2);
    gemm(stem.wb, col, stem.g, stem.b, so, nullptr, 64, 76800, 147, 1, 0);
    k_maxpool3<<<nblk((long long)64 * 120 * 160), 256, 0, stream>>>(so, rgbA, 64, 240, 320);
  }

  bf16* rgb_cur = rgbA; bf16* other = rgbB;
  int rgbC = 64, Hc = 120, Wc = 160;
  bf16* p_cur = pemb0; int pC = 8;

  // ================= downsample stages =================
  for (int i = 0; i < 4; ++i) {
    reset();
    const int Ni = NP[i], No = NP[i + 1];
    const bf16* r0pre = rgb_cur;
    int Hs = Hc, Ws = Wc;
    if (i == 1) {
      bf16* sl = (bf16*)alloc((size_t)rgbC * (Hc / 2) * (Wc / 2) * 2);
      k_slice2<<<nblk((long long)rgbC * (Hc / 2) * (Wc / 2)), 256, 0, stream>>>(
          rgb_cur, sl, rgbC, Hc, Wc);
      r0pre = sl; Hs = Hc / 2; Ws = Wc / 2;
    }
    const int hw = Hs * Ws;

    // rndla block: cat(p_emb, gather_max(p_emb, nei)) -> rndla_ds conv
    bf16* enc_in = (bf16*)alloc((size_t)2 * pC * Ni * 2);
    d2d(enc_in, p_cur, (long long)pC * Ni);
    gath(p_cur, pC, Ni, cld_nei[i], Ni, 16, enc_in + (size_t)pC * Ni);
    bf16* enc = (i == 0) ? dse0 : (bf16*)alloc((size_t)DSP[i] * Ni * 2);
    cbr(rndla_ds[i], enc_in, Ni, enc);

    // p0 (sub-sample max) -> rows [0,DSP) of point-fuse input
    bf16* pf_in = (bf16*)alloc((size_t)2 * DSP[i] * No * 2);
    gath(enc, DSP[i], Ni, cld_sub[i], No, 16, pf_in);

    // p2r path
    bf16* p2r = (bf16*)alloc((size_t)DSR[i] * No * 2);
    cbr(ds_p2r_pre[i], pf_in, No, p2r);
    bf16* fuse_in = (bf16*)alloc((size_t)2 * DSR[i] * hw * 2);
    cbr(cnn_ds[i], r0pre, hw, fuse_in);  // r0 -> rows [0,DSR)
    gath(p2r, DSR[i], No, p2r_ds[i], hw, 1, fuse_in + (size_t)DSR[i] * hw);
    cbr(ds_p2r_fuse[i], fuse_in, hw, other);  // new rgb_emb

    // r2p path (reads r0 region of fuse_in)
    bf16* r2p = (bf16*)alloc((size_t)DSR[i] * No * 2);
    gath(fuse_in, DSR[i], hw, r2p_ds[i], No, 16, r2p);
    cbr(ds_r2p_pre[i], r2p, No, pf_in + (size_t)DSP[i] * No);
    cbr(ds_r2p_fuse[i], pf_in, No, dsp[i]);  // new p_emb (persisted as ds_emb)

    bf16* t = rgb_cur; rgb_cur = other; other = t;
    rgbC = DSR[i]; Hc = Hs; Wc = Ws;
    p_cur = dsp[i]; pC = DSP[i];
  }

  // ================= upsample stages =================
  CBRp* upconv[3] = { &cnn_up1, &cnn_up2, &cnn_final };
  bf16* dsk[3]  = { dsp[2], dsp[1], dsp[0] };
  const int dskC[3]  = {256, 128, 64};
  const int decin[3] = {768, 384, 192};
  int pN = NP[4];  // point count of p_cur (50 entering)
  for (int i = 0; i < 3; ++i) {
    reset();
    const int N = NP[3 - i];
    int hw;
    const bf16* convin = rgb_cur;
    if (i < 2) {
      hw = 4 * Hc * Wc;
      bf16* u = (bf16*)alloc((size_t)rgbC * hw * 2);
      k_up2<<<nblk((long long)rgbC * hw), 256, 0, stream>>>(rgb_cur, u, rgbC, Hc, Wc);
      convin = u; Hc *= 2; Wc *= 2;
    } else {
      hw = Hc * Wc;
    }
    bf16* fuse_in = (bf16*)alloc((size_t)2 * UPR[i] * hw * 2);
    cbr(*upconv[i], convin, hw, fuse_in);  // r0 -> rows [0,UPR)

    // dec = rndla_up(cat(ds_emb_k, interp(p_emb)))
    bf16* dec_in = (bf16*)alloc((size_t)decin[i] * N * 2);
    d2d(dec_in, dsk[i], (long long)dskC[i] * N);
    gath(p_cur, pC, pN, cld_interp[3 - i], N, 1, dec_in + (size_t)dskC[i] * N);
    bf16* dec = (bf16*)alloc((size_t)UPP[i] * N * 2);
    cbr(rndla_up[i], dec_in, N, dec);

    // p2r path
    bf16* p2r = (bf16*)alloc((size_t)UPR[i] * N * 2);
    cbr(up_p2r_pre[i], dec, N, p2r);
    gath(p2r, UPR[i], N, p2r_up[i], hw, 1, fuse_in + (size_t)UPR[i] * hw);
    cbr(up_p2r_fuse[i], fuse_in, hw, other);  // new rgb_emb

    // r2p path
    bf16* r2p = (bf16*)alloc((size_t)UPR[i] * N * 2);
    gath(fuse_in, UPR[i], hw, r2p_up[i], N, 16, r2p);
    bf16* pf_in = (bf16*)alloc((size_t)2 * UPP[i] * N * 2);
    d2d(pf_in, dec, (long long)UPP[i] * N);
    cbr(up_r2p_pre[i], r2p, N, pf_in + (size_t)UPP[i] * N);
    cbr(up_r2p_fuse[i], pf_in, N, upp[i]);

    bf16* t = rgb_cur; rgb_cur = other; other = t;
    rgbC = UPR[i]; p_cur = upp[i]; pC = UPP[i]; pN = N;
  }

  // ================= final stage + heads =================
  reset();
  bf16* u  = (bf16*)alloc((size_t)64 * 307200 * 2);
  k_up2<<<nblk((long long)64 * 307200), 256, 0, stream>>>(rgb_cur, u, 64, Hc, Wc);
  bf16* t1 = (bf16*)alloc((size_t)64 * 307200 * 2);
  cbr(cnn_up3, u, 307200, t1);
  cbr(cnn_final, t1, 307200, other);
  bf16* rgbF = other;

  bf16* fin = (bf16*)alloc((size_t)128 * 12800 * 2);
  d2d(fin, dse0, (long long)64 * 12800);
  gath(p_cur, 64, 3200, cld_interp[0], 12800, 1, fin + (size_t)64 * 12800);
  cbr(rndla_up[3], fin, 12800, pembF);

  bf16* rgbd = (bf16*)alloc((size_t)128 * 12800 * 2);
  gath(rgbF, 64, 307200, choose, 12800, 1, rgbd);
  d2d(rgbd + (size_t)64 * 12800, pembF, (long long)64 * 12800);

  float* outp = (float*)d_out;
  auto run_head = [&](CBRp* hs, const float* bias, bf16* wb, int O, float* dst, int layout) {
    bf16* h1 = (bf16*)alloc((size_t)128 * 12800 * 2);
    bf16* h2 = (bf16*)alloc((size_t)128 * 12800 * 2);
    cbr(hs[0], rgbd, 12800, h1);
    cbr(hs[1], h1, 12800, h2);
    cbr(hs[2], h2, 12800, h1);
    gemm(wb, h1, nullptr, bias, nullptr, dst, O, 12800, 128, 0, layout);
  };
  run_head(seg_h, seg_bias, seg_wb, 22, outp, 1);                       // [22,12800]
  run_head(kp_h,  kp_bias,  kp_wb,  24, outp + (size_t)22 * 12800, 2);  // (8,12800,3)
  run_head(ctr_h, ctr_bias, ctr_wb, 3,  outp + (size_t)46 * 12800, 2);  // (1,12800,3)
}
